// DGCNN_partseg_19207093748092
// MI455X (gfx1250) — compile-verified, matrix operand order checked by reading, not run on hardware
//
#include <hip/hip_runtime.h>
#include <math.h>

// ---------------- constants ----------------
#define BATCH 8
#define NPTS  2048

typedef __bf16 bf16;
typedef __attribute__((ext_vector_type(16))) __bf16 v16bf;
typedef __attribute__((ext_vector_type(8)))  float  v8f;

#define DEVINL __device__ __forceinline__

DEVINL float leakyf(float x) { return x > 0.f ? x : 0.2f * x; }

// ---------------- WMMA fragment helpers (ISA 7.12.2, 16-bit 16x32 A / 32x16 B) ----------------
DEVINL int frag_k(int lane, int e) {
    int k = (e < 8) ? e : e + 8;
    return k + ((lane & 16) ? 8 : 0);
}
// As: [m][k] row-major tile (16 x 32), ld = row stride
DEVINL v16bf frag_A_lds(const float* As, int ld, int lane) {
    v16bf a; int m = lane & 15;
#pragma unroll
    for (int e = 0; e < 16; ++e) a[e] = (bf16)As[m * ld + frag_k(lane, e)];
    return a;
}
// Bs: [k][n] row-major tile (32 x 16..), ld = row stride
DEVINL v16bf frag_B_lds(const float* Bs, int ld, int lane) {
    v16bf b; int n = lane & 15;
#pragma unroll
    for (int e = 0; e < 16; ++e) b[e] = (bf16)Bs[frag_k(lane, e) * ld + n];
    return b;
}

// ---------------- generic 1x1-conv GEMM:  Y[b,o,p] = act((W[o,:]·X[b,:,p])*S[o]+Bb[o]) ----------------
// X: (B, C, P), W: (O, C), Y: (B, O, P).  Block = 256 thr (8 waves, 2M x 4N of 16x16 tiles).
__global__ void gemm_conv_kernel(const float* __restrict__ W, const float* __restrict__ S,
                                 const float* __restrict__ Bb, const float* __restrict__ X,
                                 float* __restrict__ Y, int O, int C, long long P, int act) {
    __shared__ float As[32 * 32];
    __shared__ float Bs[32 * 64];
    int b = blockIdx.z;
    long long p0 = (long long)blockIdx.x * 64;
    int o0 = blockIdx.y * 32;
    int tid = threadIdx.x, lane = tid & 31, w = tid >> 5;
    int wm = w & 1, wn = w >> 1;
    const float* Xb = X + (long long)b * C * P;
    v8f acc = {};
    for (int c0 = 0; c0 < C; c0 += 32) {
        for (int e = tid; e < 32 * 32; e += 256) {
            int mo = e >> 5, kc = e & 31;
            int o = o0 + mo, c = c0 + kc;
            As[e] = (o < O && c < C) ? W[(long long)o * C + c] : 0.f;
        }
        for (int e = tid; e < 32 * 64; e += 256) {
            int kc = e >> 6, pn = e & 63;
            int c = c0 + kc; long long p = p0 + pn;
            Bs[e] = (c < C && p < P) ? Xb[(long long)c * P + p] : 0.f;
        }
        __syncthreads();
        v16bf a = frag_A_lds(As + wm * 16 * 32, 32, lane);
        v16bf bfr = frag_B_lds(Bs + wn * 16, 64, lane);
        acc = __builtin_amdgcn_wmma_f32_16x16x32_bf16(false, a, false, bfr, (short)0, acc, false, false);
        __syncthreads();
    }
    int n = lane & 15, mb = (lane & 16) ? 8 : 0;
    long long p = p0 + wn * 16 + n;
    if (p < P) {
#pragma unroll
        for (int r = 0; r < 8; ++r) {
            int o = o0 + wm * 16 + mb + r;
            if (o < O) {
                float y = acc[r];
                if (S) y = y * S[o] + Bb[o];
                if (act) y = leakyf(y);
                Y[((long long)b * O + o) * P + p] = y;
            }
        }
    }
}

// ---------------- fused conv(O=128) + LeakyReLU + max over neighbor dim k ----------------
// H: (B, C, Npts, k) -> out: (B, 128, Npts).  Block = 256 thr; wave w owns out channels [16w,16w+16).
template <int C>
__global__ void convmax_kernel(const float* __restrict__ W, const float* __restrict__ S,
                               const float* __restrict__ Bb, const float* __restrict__ H,
                               float* __restrict__ out, int Npts, int k) {
    constexpr int NST = C / 32;
    __shared__ float As[128 * 32];
    __shared__ float Bs[32 * 16];
    int b = blockIdx.y, i0 = blockIdx.x * 16;
    int tid = threadIdx.x, lane = tid & 31, w = tid >> 5;
    v16bf afr[NST];
#pragma unroll
    for (int cs = 0; cs < NST; ++cs) {
        __syncthreads();
        for (int e = tid; e < 128 * 32; e += 256) {
            int mo = e >> 5, kc = e & 31;
            As[e] = W[(long long)mo * C + cs * 32 + kc];
        }
        __syncthreads();
        afr[cs] = frag_A_lds(As + w * 16 * 32, 32, lane);
    }
    const float* Hb = H + (long long)b * C * Npts * k;
    v8f best;
#pragma unroll
    for (int r = 0; r < 8; ++r) best[r] = -3.402823e38f;
    int mb = (lane & 16) ? 8 : 0;
    for (int j = 0; j < k; ++j) {
        v8f acc = {};
#pragma unroll
        for (int cs = 0; cs < NST; ++cs) {
            __syncthreads();
            for (int e = tid; e < 32 * 16; e += 256) {
                int kc = e >> 4, pn = e & 15;
                int i = i0 + pn;
                Bs[e] = (i < Npts) ? Hb[((long long)(cs * 32 + kc) * Npts + i) * k + j] : 0.f;
            }
            __syncthreads();
            acc = __builtin_amdgcn_wmma_f32_16x16x32_bf16(false, afr[cs], false,
                                                          frag_B_lds(Bs, 16, lane),
                                                          (short)0, acc, false, false);
        }
#pragma unroll
        for (int r = 0; r < 8; ++r) {
            int o = w * 16 + mb + r;
            float y = leakyf(acc[r] * S[o] + Bb[o]);
            best[r] = fmaxf(best[r], y);
        }
    }
    int n = lane & 15;
    int i = i0 + n;
    if (i < Npts) {
#pragma unroll
        for (int r = 0; r < 8; ++r) {
            int o = w * 16 + mb + r;
            out[((long long)b * 128 + o) * Npts + i] = best[r];
        }
    }
}

// ---------------- kNN: per-query top-k smallest squared distance ----------------
// q: (B,C,Nq), r: (B,C,Nr) -> idx: (B,Nq,k) sorted ascending by distance.
__global__ void knn_kernel(const float* __restrict__ q, const float* __restrict__ r,
                           int* __restrict__ idx, int C, int Nq, int Nr, int k) {
    extern __shared__ float rlds[];  // [C][128]
    const int TILE = 128;
    int b = blockIdx.y;
    int qi = blockIdx.x * 128 + threadIdx.x;
    float bestd[32]; int besti[32];
    for (int t = 0; t < 32; ++t) { bestd[t] = 3.402823e38f; besti[t] = 0; }
    float curmax = 3.402823e38f; int curpos = 0;
    const float* qb = q + (long long)b * C * Nq;
    const float* rb = r + (long long)b * C * Nr;
    for (int r0 = 0; r0 < Nr; r0 += TILE) {
        int tn = min(TILE, Nr - r0);
        __syncthreads();
        for (int e = threadIdx.x; e < C * tn; e += blockDim.x) {
            int c = e / tn, rl = e % tn;
            rlds[c * TILE + rl] = rb[(long long)c * Nr + r0 + rl];
        }
        __syncthreads();
        if (qi < Nq) {
            for (int rl = 0; rl < tn; ++rl) {
                float d = 0.f;
                for (int c = 0; c < C; ++c) {
                    float df = qb[(long long)c * Nq + qi] - rlds[c * TILE + rl];
                    d += df * df;
                }
                if (d < curmax) {
                    bestd[curpos] = d; besti[curpos] = r0 + rl;
                    curmax = -3.402823e38f;
                    for (int t = 0; t < k; ++t)
                        if (bestd[t] > curmax) { curmax = bestd[t]; curpos = t; }
                }
            }
        }
    }
    if (qi < Nq) {
        for (int a = 0; a < k; ++a) {  // selection sort ascending
            int mp = a;
            for (int t = a + 1; t < k; ++t) if (bestd[t] < bestd[mp]) mp = t;
            float td = bestd[a]; bestd[a] = bestd[mp]; bestd[mp] = td;
            int ti = besti[a]; besti[a] = besti[mp]; besti[mp] = ti;
            idx[((long long)b * Nq + qi) * k + a] = besti[a];
        }
    }
}

// ---------------- edge features: ef[b, 0:C, i, j]=nbr-ctr ; ef[b, C:2C, i, j]=ctr ----------------
__global__ void edgefeat_kernel(const float* __restrict__ feat, const int* __restrict__ idx,
                                float* __restrict__ ef, int C, int Npts, int k) {
    long long t = (long long)blockIdx.x * blockDim.x + threadIdx.x;
    long long total = (long long)BATCH * Npts * k;
    if (t >= total) return;
    int j = (int)(t % k); long long t2 = t / k;
    int i = (int)(t2 % Npts); int b = (int)(t2 / Npts);
    int nb = idx[((long long)b * Npts + i) * k + j];
    const float* fb = feat + (long long)b * C * Npts;
    float* eb = ef + (long long)b * (2 * C) * Npts * k;
    for (int c = 0; c < C; ++c) {
        float ctr = fb[(long long)c * Npts + i];
        float nbr = fb[(long long)c * Npts + nb];
        eb[((long long)c * Npts + i) * k + j] = nbr - ctr;
        eb[((long long)(C + c) * Npts + i) * k + j] = ctr;
    }
}

// ---------------- SAGPool score: s[b,i] = ws · feat[b,:,i]  (C=128 fixed) ----------------
__global__ void score_kernel(const float* __restrict__ ws, const float* __restrict__ feat,
                             float* __restrict__ s, int Npts) {
    int t = blockIdx.x * blockDim.x + threadIdx.x;
    if (t >= BATCH * Npts) return;
    int i = t % Npts, b = t / Npts;
    const float* fb = feat + (long long)b * 128 * Npts;
    float acc = 0.f;
    for (int c = 0; c < 128; ++c) acc += ws[c] * fb[(long long)c * Npts + i];
    s[t] = acc;
}

// ---------------- neighborhood mean smoothing of scores ----------------
__global__ void smooth_kernel(const float* __restrict__ s, const int* __restrict__ idx,
                              float* __restrict__ out, int Npts, int kfull, int kuse) {
    int t = blockIdx.x * blockDim.x + threadIdx.x;
    if (t >= BATCH * Npts) return;
    int i = t % Npts, b = t / Npts;
    const int* ib = idx + ((long long)b * Npts + i) * kfull;
    float acc = 0.f;
    for (int j = 0; j < kuse; ++j) acc += s[b * Npts + ib[j]];
    out[t] = acc / (float)kuse;
}

// ---------------- iterative top-k selection (values desc, ties -> lower index) ----------------
__global__ void topk_kernel(const float* __restrict__ s, float* __restrict__ vals,
                            int* __restrict__ idxo, int Npts, int nout) {
    __shared__ float sc[2048];
    __shared__ float rv[256];
    __shared__ int   ri[256];
    int b = blockIdx.x, tid = threadIdx.x;
    for (int i = tid; i < Npts; i += 256) sc[i] = s[b * Npts + i];
    __syncthreads();
    for (int t = 0; t < nout; ++t) {
        float bv = -3.402823e38f; int bi = 1 << 30;
        for (int i = tid; i < Npts; i += 256)
            if (sc[i] > bv || (sc[i] == bv && i < bi)) { bv = sc[i]; bi = i; }
        rv[tid] = bv; ri[tid] = bi;
        __syncthreads();
        if (tid == 0) {
            for (int u = 1; u < 256; ++u)
                if (rv[u] > rv[0] || (rv[u] == rv[0] && ri[u] < ri[0])) { rv[0] = rv[u]; ri[0] = ri[u]; }
            vals[b * nout + t] = rv[0];
            idxo[b * nout + t] = ri[0];
            sc[ri[0]] = -3.402823e38f;
        }
        __syncthreads();
    }
}

// ---------------- gather with optional tanh(value) gating ----------------
// out[b*obs + c*M + m] = feat[(b*C+c)*Npts + idx[b*M+m]] * (use_tanh ? tanh(vals[b*M+m]) : 1)
__global__ void gather_feat_kernel(const float* __restrict__ feat, const int* __restrict__ idx,
                                   const float* __restrict__ vals, float* __restrict__ out,
                                   long long obs, int C, int Npts, int M, int use_tanh) {
    long long t = (long long)blockIdx.x * blockDim.x + threadIdx.x;
    long long total = (long long)BATCH * C * M;
    if (t >= total) return;
    int m = (int)(t % M); long long t2 = t / M;
    int c = (int)(t2 % C); int b = (int)(t2 / C);
    int src = idx[(long long)b * M + m];
    float v = feat[((long long)b * C + c) * Npts + src];
    if (use_tanh) v *= tanhf(vals[(long long)b * M + m]);
    out[(long long)b * obs + (long long)c * M + m] = v;
}

// ---------------- aggregate: gather + max over k ----------------
__global__ void gathermax_kernel(const float* __restrict__ feat, const int* __restrict__ idx,
                                 float* __restrict__ out, long long obs, int C, int Npts,
                                 int M, int k) {
    long long t = (long long)blockIdx.x * blockDim.x + threadIdx.x;
    long long total = (long long)BATCH * C * M;
    if (t >= total) return;
    int m = (int)(t % M); long long t2 = t / M;
    int c = (int)(t2 % C); int b = (int)(t2 / C);
    const int* ib = idx + ((long long)b * M + m) * k;
    const float* fb = feat + ((long long)b * C + c) * Npts;
    float v = -3.402823e38f;
    for (int j = 0; j < k; ++j) v = fmaxf(v, fb[ib[j]]);
    out[(long long)b * obs + (long long)c * M + m] = v;
}

// ---------------- copy channels into a concat buffer ----------------
__global__ void copych_kernel(const float* __restrict__ in, float* __restrict__ out,
                              long long obs, int C, long long P) {
    long long t = (long long)blockIdx.x * blockDim.x + threadIdx.x;
    long long total = (long long)BATCH * C * P;
    if (t >= total) return;
    long long p = t % P; long long t2 = t / P;
    int c = (int)(t2 % C); int b = (int)(t2 / C);
    out[(long long)b * obs + (long long)c * P + p] = in[((long long)b * C + c) * P + p];
}

// ---------------- broadcast per-batch vector over P points ----------------
__global__ void bcast_kernel(const float* __restrict__ g, float* __restrict__ out,
                             long long obs, int C, int P) {
    long long t = (long long)blockIdx.x * blockDim.x + threadIdx.x;
    long long total = (long long)BATCH * C * P;
    if (t >= total) return;
    int p = (int)(t % P); long long t2 = t / P;
    int c = (int)(t2 % C); int b = (int)(t2 / C);
    out[(long long)b * obs + (long long)c * P + p] = g[(long long)b * C + c];
}

// ---------------- small per-batch head GEMM (N = BATCH columns) ----------------
__global__ void headgemm_kernel(const float* __restrict__ W, const float* __restrict__ S,
                                const float* __restrict__ Bb, const float* __restrict__ X,
                                float* __restrict__ Y, long long xbs, long long ybs,
                                int O, int K, int act) {
    int t = blockIdx.x * blockDim.x + threadIdx.x;
    if (t >= BATCH * O) return;
    int o = t % O, b = t / O;
    const float* xb = X + (long long)b * xbs;
    const float* wr = W + (long long)o * K;
    float acc = 0.f;
    for (int c = 0; c < K; ++c) acc += wr[c] * xb[c];
    acc = acc * S[o] + Bb[o];
    if (act) acc = leakyf(acc);
    Y[(long long)b * ybs + o] = acc;
}

// ---------------- residual + leaky: y = leaky(y + a) ----------------
__global__ void resleaky_kernel(float* __restrict__ y, const float* __restrict__ a,
                                long long total) {
    long long t = (long long)blockIdx.x * blockDim.x + threadIdx.x;
    if (t >= total) return;
    y[t] = leakyf(y[t] + a[t]);
}

// =====================================================================
//                         host-side orchestration
// =====================================================================

// ---- int workspace offsets (in int elements) ----
#define I_IDX1 0LL
#define I_I1   524288LL
#define I_I2   528384LL
#define I_I3   529408LL
#define I_AGG1 529664LL
#define I_FID2 595200LL
#define I_NID2 660736LL
#define I_AGG2 693504LL
#define I_FID3 701696LL
#define I_NID3 709888LL
#define I_AGG3 713984LL
#define I_FID4 715008LL
#define I_UP3  716032LL
#define I_UP2  717056LL
#define I_UP1  721152LL
#define INT_TOTAL 737536LL

// ---- float workspace offsets (in float elements, after int region) ----
#define F_EF1   0LL
#define F_X1    3145728LL
#define F_SBUF  5242880LL
#define F_V1    5259264LL
#define F_V2    5263360LL
#define F_V3    5264384LL
#define F_CAT1  5264640LL
#define F_XP1   6313216LL
#define F_X2    6837504LL
#define F_CAT2  7361792LL
#define F_XP2   7623936LL
#define F_X3    7755008LL
#define F_CAT3  7886080LL
#define F_XP3   7951616LL
#define F_X4    7984384LL
#define F_GCAT  8017152LL
#define F_GOUT  8025856LL
#define F_CAT9  8027904LL
#define F_H9    8126208LL
#define F_CAT10 8191744LL
#define F_H10   8584960LL
#define F_CAT11 8847104LL
#define F_H11   10419968LL
#define F_CAT12 11468544LL
#define F_H12   17760000LL
#define F_BIG   19857152LL      /* 33,554,432 floats, reused per stage */

// ---- d_out offsets (floats) ----
#define O_OUT  0LL
#define O_RET1 819200LL
#define O_RET2 835584LL
#define O_RET3 839680LL
#define O_N1   840704LL
#define O_N2   852992LL
#define O_N3   856064LL
#define O_N1S  856832LL
#define O_N2S  869120LL
#define O_N3S  872192LL

extern "C" void kernel_launch(void* const* d_in, const int* in_sizes, int n_in,
                              void* d_out, int out_size, void* d_ws, size_t ws_size,
                              hipStream_t stream) {
    (void)in_sizes; (void)n_in; (void)out_size; (void)ws_size;
    const float* x = (const float*)d_in[0];   // (8,3,2048)
    const float* l = (const float*)d_in[1];   // (8,16)
    // conv params: 0=c1,1=c2,2=c1p,3=c3,4=c4,5=c2p,6=c5,7=c3p,8=c6,9=c6m,10=c7,11=c8,12=c9,13=c10,14=c11,15=c12
    auto Wp = [&](int ci) { return (const float*)d_in[2 + 3 * ci]; };
    auto Sp = [&](int ci) { return (const float*)d_in[3 + 3 * ci]; };
    auto Bp = [&](int ci) { return (const float*)d_in[4 + 3 * ci]; };
    const float* c13w = (const float*)d_in[50];
    const float* ws1 = (const float*)d_in[51];
    const float* ws2 = (const float*)d_in[52];
    const float* ws3 = (const float*)d_in[53];

    int* ib = (int*)d_ws;
    float* fbb = (float*)((char*)d_ws + (size_t)INT_TOTAL * 4);
    float* out = (float*)d_out;

    auto gr = [](long long n) { return dim3((unsigned)((n + 255) / 256)); };
    auto knn = [&](const float* q, const float* r, int* idx, int C, int Nq, int Nr, int k) {
        dim3 g((Nq + 127) / 128, BATCH);
        knn_kernel<<<g, 128, (size_t)C * 128 * 4, stream>>>(q, r, idx, C, Nq, Nr, k);
    };
    auto gemm = [&](const float* W, const float* S, const float* Bb, const float* X, float* Y,
                    int O, int C, long long P, int act) {
        dim3 g((unsigned)((P + 63) / 64), (O + 31) / 32, BATCH);
        gemm_conv_kernel<<<g, 256, 0, stream>>>(W, S, Bb, X, Y, O, C, P, act);
    };

    // ===== Stage 1: graph_feature(x,32) -> c1 -> c2 -> max -> x1 =====
    knn(x, x, ib + I_IDX1, 3, 2048, 2048, 32);
    edgefeat_kernel<<<gr(8LL * 2048 * 32), 256, 0, stream>>>(x, ib + I_IDX1, fbb + F_EF1, 3, 2048, 32);
    gemm(Wp(0), Sp(0), Bp(0), fbb + F_EF1, fbb + F_BIG, 64, 6, 65536, 1);       // h1 (B,64,2048,32)
    {
        dim3 g(2048 / 16, BATCH);
        convmax_kernel<64><<<g, 256, 0, stream>>>(Wp(1), Sp(1), Bp(1), fbb + F_BIG, fbb + F_X1, 2048, 32);
    }
    // ===== index_select #1 (n_out=512, neighs=16) =====
    score_kernel<<<gr(8LL * 2048), 256, 0, stream>>>(ws1, fbb + F_X1, fbb + F_SBUF, 2048);
    smooth_kernel<<<gr(8LL * 2048), 256, 0, stream>>>(fbb + F_SBUF, ib + I_IDX1, out + O_RET1, 2048, 32, 16);
    topk_kernel<<<BATCH, 256, 0, stream>>>(out + O_RET1, fbb + F_V1, ib + I_I1, 2048, 512);
    gather_feat_kernel<<<gr(8LL * 3 * 512), 256, 0, stream>>>(x, ib + I_I1, nullptr, out + O_N1S,
                                                              3LL * 512, 3, 2048, 512, 0);
    hipMemcpyAsync(out + O_N1, out + O_N1S, 12288 * 4, hipMemcpyDeviceToDevice, stream);
    gather_feat_kernel<<<gr(8LL * 128 * 512), 256, 0, stream>>>(fbb + F_X1, ib + I_I1, fbb + F_V1,
                                                                fbb + F_CAT1, 256LL * 512, 128, 2048, 512, 1);
    knn(out + O_N1S, x, ib + I_AGG1, 3, 512, 2048, 16);
    gathermax_kernel<<<gr(8LL * 128 * 512), 256, 0, stream>>>(fbb + F_X1, ib + I_AGG1,
                                                              fbb + F_CAT1 + 128LL * 512, 256LL * 512,
                                                              128, 2048, 512, 16);
    gemm(Wp(2), Sp(2), Bp(2), fbb + F_CAT1, fbb + F_XP1, 128, 256, 512, 1);     // xp1 (B,128,512)

    // ===== Stage 2: graph_feature(xp1,16) -> c3 -> c4 -> max -> +xp1 =====
    knn(fbb + F_XP1, fbb + F_XP1, ib + I_FID2, 128, 512, 512, 16);
    edgefeat_kernel<<<gr(8LL * 512 * 16), 256, 0, stream>>>(fbb + F_XP1, ib + I_FID2, fbb + F_BIG, 128, 512, 16);
    gemm(Wp(3), Sp(3), Bp(3), fbb + F_BIG, fbb + F_BIG + 16777216LL, 128, 256, 8192, 1);  // h2
    {
        dim3 g(512 / 16, BATCH);
        convmax_kernel<128><<<g, 256, 0, stream>>>(Wp(4), Sp(4), Bp(4), fbb + F_BIG + 16777216LL,
                                                   fbb + F_X2, 512, 16);
    }
    resleaky_kernel<<<gr(8LL * 128 * 512), 256, 0, stream>>>(fbb + F_X2, fbb + F_XP1, 8LL * 128 * 512);
    // ===== index_select #2 (n_out=128, neighs=8) =====
    score_kernel<<<gr(8LL * 512), 256, 0, stream>>>(ws2, fbb + F_X2, fbb + F_SBUF, 512);
    knn(out + O_N1S, out + O_N1S, ib + I_NID2, 3, 512, 512, 8);
    smooth_kernel<<<gr(8LL * 512), 256, 0, stream>>>(fbb + F_SBUF, ib + I_NID2, out + O_RET2, 512, 8, 8);
    topk_kernel<<<BATCH, 256, 0, stream>>>(out + O_RET2, fbb + F_V2, ib + I_I2, 512, 128);
    gather_feat_kernel<<<gr(8LL * 3 * 128), 256, 0, stream>>>(out + O_N1S, ib + I_I2, nullptr, out + O_N2S,
                                                              3LL * 128, 3, 512, 128, 0);
    hipMemcpyAsync(out + O_N2, out + O_N2S, 3072 * 4, hipMemcpyDeviceToDevice, stream);
    gather_feat_kernel<<<gr(8LL * 128 * 128), 256, 0, stream>>>(fbb + F_X2, ib + I_I2, fbb + F_V2,
                                                                fbb + F_CAT2, 256LL * 128, 128, 512, 128, 1);
    knn(out + O_N2S, out + O_N1S, ib + I_AGG2, 3, 128, 512, 8);
    gathermax_kernel<<<gr(8LL * 128 * 128), 256, 0, stream>>>(fbb + F_X2, ib + I_AGG2,
                                                              fbb + F_CAT2 + 128LL * 128, 256LL * 128,
                                                              128, 512, 128, 8);
    gemm(Wp(5), Sp(5), Bp(5), fbb + F_CAT2, fbb + F_XP2, 128, 256, 128, 1);     // xp2 (B,128,128)

    // ===== Stage 3: graph_feature(xp2,8) -> c5 -> max -> +xp2 =====
    knn(fbb + F_XP2, fbb + F_XP2, ib + I_FID3, 128, 128, 128, 8);
    edgefeat_kernel<<<gr(8LL * 128 * 8), 256, 0, stream>>>(fbb + F_XP2, ib + I_FID3, fbb + F_BIG, 128, 128, 8);
    {
        dim3 g(128 / 16, BATCH);
        convmax_kernel<256><<<g, 256, 0, stream>>>(Wp(6), Sp(6), Bp(6), fbb + F_BIG, fbb + F_X3, 128, 8);
    }
    resleaky_kernel<<<gr(8LL * 128 * 128), 256, 0, stream>>>(fbb + F_X3, fbb + F_XP2, 8LL * 128 * 128);
    // ===== index_select #3 (n_out=32, neighs=4) =====
    score_kernel<<<gr(8LL * 128), 256, 0, stream>>>(ws3, fbb + F_X3, fbb + F_SBUF, 128);
    knn(out + O_N2S, out + O_N2S, ib + I_NID3, 3, 128, 128, 4);
    smooth_kernel<<<gr(8LL * 128), 256, 0, stream>>>(fbb + F_SBUF, ib + I_NID3, out + O_RET3, 128, 4, 4);
    topk_kernel<<<BATCH, 256, 0, stream>>>(out + O_RET3, fbb + F_V3, ib + I_I3, 128, 32);
    gather_feat_kernel<<<gr(8LL * 3 * 32), 256, 0, stream>>>(out + O_N2S, ib + I_I3, nullptr, out + O_N3S,
                                                             3LL * 32, 3, 128, 32, 0);
    hipMemcpyAsync(out + O_N3, out + O_N3S, 768 * 4, hipMemcpyDeviceToDevice, stream);
    gather_feat_kernel<<<gr(8LL * 128 * 32), 256, 0, stream>>>(fbb + F_X3, ib + I_I3, fbb + F_V3,
                                                               fbb + F_CAT3, 256LL * 32, 128, 128, 32, 1);
    knn(out + O_N3S, out + O_N2S, ib + I_AGG3, 3, 32, 128, 4);
    gathermax_kernel<<<gr(8LL * 128 * 32), 256, 0, stream>>>(fbb + F_X3, ib + I_AGG3,
                                                             fbb + F_CAT3 + 128LL * 32, 256LL * 32,
                                                             128, 128, 32, 4);
    gemm(Wp(7), Sp(7), Bp(7), fbb + F_CAT3, fbb + F_XP3, 128, 256, 32, 1);      // xp3 (B,128,32)

    // ===== Stage 4: graph_feature(xp3,4) -> c6 -> max -> +xp3 =====
    knn(fbb + F_XP3, fbb + F_XP3, ib + I_FID4, 128, 32, 32, 4);
    edgefeat_kernel<<<gr(8LL * 32 * 4), 256, 0, stream>>>(fbb + F_XP3, ib + I_FID4, fbb + F_BIG, 128, 32, 4);
    {
        dim3 g(2, BATCH);
        convmax_kernel<256><<<g, 256, 0, stream>>>(Wp(8), Sp(8), Bp(8), fbb + F_BIG, fbb + F_X4, 32, 4);
    }
    resleaky_kernel<<<gr(8LL * 128 * 32), 256, 0, stream>>>(fbb + F_X4, fbb + F_XP3, 8LL * 128 * 32);

    // ===== Global head: c6m -> (cat l->c7) -> c8 -> broadcast =====
    headgemm_kernel<<<gr(8LL * 1024), 256, 0, stream>>>(Wp(9), Sp(9), Bp(9), fbb + F_X4, fbb + F_GCAT,
                                                        4096, 1088, 1024, 4096, 1);
    headgemm_kernel<<<gr(8LL * 64), 256, 0, stream>>>(Wp(10), Sp(10), Bp(10), l, fbb + F_GCAT + 1024,
                                                      16, 1088, 64, 16, 1);
    headgemm_kernel<<<gr(8LL * 256), 256, 0, stream>>>(Wp(11), Sp(11), Bp(11), fbb + F_GCAT, fbb + F_GOUT,
                                                       1088, 256, 256, 1088, 1);
    bcast_kernel<<<gr(8LL * 256 * 32), 256, 0, stream>>>(fbb + F_GOUT, fbb + F_CAT9, 384LL * 32, 256, 32);
    copych_kernel<<<gr(8LL * 128 * 32), 256, 0, stream>>>(fbb + F_X4, fbb + F_CAT9 + 256LL * 32,
                                                          384LL * 32, 128, 32);
    gemm(Wp(12), Sp(12), Bp(12), fbb + F_CAT9, fbb + F_H9, 256, 384, 32, 1);    // h9 (B,256,32)

    // ===== Decoder: unpool -> c10 -> unpool -> c11 -> unpool -> c12 -> c13 =====
    knn(out + O_N2S, out + O_N3S, ib + I_UP3, 3, 128, 32, 1);
    gather_feat_kernel<<<gr(8LL * 256 * 128), 256, 0, stream>>>(fbb + F_H9, ib + I_UP3, nullptr,
                                                                fbb + F_CAT10, 384LL * 128, 256, 32, 128, 0);
    copych_kernel<<<gr(8LL * 128 * 128), 256, 0, stream>>>(fbb + F_X3, fbb + F_CAT10 + 256LL * 128,
                                                           384LL * 128, 128, 128);
    gemm(Wp(13), Sp(13), Bp(13), fbb + F_CAT10, fbb + F_H10, 256, 384, 128, 1); // h10 (B,256,128)

    knn(out + O_N1S, out + O_N2S, ib + I_UP2, 3, 512, 128, 1);
    gather_feat_kernel<<<gr(8LL * 256 * 512), 256, 0, stream>>>(fbb + F_H10, ib + I_UP2, nullptr,
                                                                fbb + F_CAT11, 384LL * 512, 256, 128, 512, 0);
    copych_kernel<<<gr(8LL * 128 * 512), 256, 0, stream>>>(fbb + F_X2, fbb + F_CAT11 + 256LL * 512,
                                                           384LL * 512, 128, 512);
    gemm(Wp(14), Sp(14), Bp(14), fbb + F_CAT11, fbb + F_H11, 256, 384, 512, 1); // h11 (B,256,512)

    knn(x, out + O_N1S, ib + I_UP1, 3, 2048, 512, 1);
    gather_feat_kernel<<<gr(8LL * 256 * 2048), 256, 0, stream>>>(fbb + F_H11, ib + I_UP1, nullptr,
                                                                 fbb + F_CAT12, 384LL * 2048, 256, 512, 2048, 0);
    copych_kernel<<<gr(8LL * 128 * 2048), 256, 0, stream>>>(fbb + F_X1, fbb + F_CAT12 + 256LL * 2048,
                                                            384LL * 2048, 128, 2048);
    gemm(Wp(15), Sp(15), Bp(15), fbb + F_CAT12, fbb + F_H12, 128, 384, 2048, 1); // h12 (B,128,2048)

    gemm(c13w, nullptr, nullptr, fbb + F_H12, out + O_OUT, 50, 128, 2048, 0);    // out (B,50,2048)
}
